// NeuralMemory_16329465660007
// MI455X (gfx1250) — matrix operand order repaired and checked by reading, not compiled, not run on hardware
//
#include <hip/hip_runtime.h>
#include <hip/hip_bf16.h>

typedef __attribute__((ext_vector_type(16))) _Float16 v16h;
typedef __attribute__((ext_vector_type(8)))  _Float16 v8h;
typedef __attribute__((ext_vector_type(4)))  _Float16 v4h;
typedef __attribute__((ext_vector_type(8)))  float    v8f;
typedef __attribute__((ext_vector_type(4)))  float    v4f;

#define BATCH  8
#define SEQ    8192
#define DIM    128
#define MEMD   128
#define ST     64
#define NCHUNK (SEQ / ST)

// ---------------------------------------------------------------------------
// WMMA fragment loaders (CDNA5 ISA 7.12.2 layouts, wave32)
// ---------------------------------------------------------------------------

// A fragment: 16x32 f16 tile, row-major source with row stride `ld` (elems).
__device__ __forceinline__ v16h load_frag_a(const _Float16* p, int ld, int lane) {
  const int m  = lane & 15;
  const int kb = (lane >> 4) << 3;          // 0 or 8
  const _Float16* row = p + m * ld;
  v8h lo = *(const v8h*)(row + kb);
  v8h hi = *(const v8h*)(row + 16 + kb);
  v16h a;
#pragma unroll
  for (int i = 0; i < 8; ++i) { a[i] = lo[i]; a[i + 8] = hi[i]; }
  return a;
}

// B fragment (32x16) loaded from B-transposed storage: row-major [N][K].
__device__ __forceinline__ v16h load_frag_bT(const _Float16* pT, int ld, int lane) {
  const int n  = lane & 15;
  const int kb = (lane >> 4) << 3;          // 0 or 8
  const _Float16* row = pT + n * ld + kb;
  v8h lo = *(const v8h*)(row);
  v8h hi = *(const v8h*)(row + 16);
  v16h b;
#pragma unroll
  for (int i = 0; i < 8; ++i) { b[i] = lo[i]; b[i + 8] = hi[i]; }
  return b;
}

// ---------------------------------------------------------------------------
// Fused neural-memory scan: one block per batch.
//   - M^T resident in LDS (f32 master + f16 mirror) for all 128 chunks
//   - all four weight matrices staged in LDS as f16 at prologue
//   - next x chunk prefetched (global_prefetch_b8) while current computes
// ---------------------------------------------------------------------------
__global__ __launch_bounds__(256)
void nm_scan_kernel(const float* __restrict__ x,     // [B,S,D] f32
                    const float* __restrict__ Wq,    // [D,D]
                    const float* __restrict__ Wk,    // [D,D]
                    const float* __restrict__ Wv,    // [MEM,D]
                    const float* __restrict__ Wg_w,  // [D]
                    const float* __restrict__ Wg_b,  // [1]
                    const float* __restrict__ Wout,  // [D,MEM]
                    float*       __restrict__ out)   // [B,S,D] f32
{
  extern __shared__ __align__(16) char smem[];
  float*    MT32 = (float*)    (smem);            // [MEM][DIM] f32 M^T        64KB
  _Float16* MT16 = (_Float16*) (smem +  65536);   // [MEM][DIM] f16 mirror     32KB
  _Float16* XL   = (_Float16*) (smem +  98304);   // [ST][DIM]  x chunk        16KB
  _Float16* QL   = (_Float16*) (smem + 114688);   // [ST][DIM]  q chunk        16KB
  _Float16* KGT  = (_Float16*) (smem + 131072);   // [DIM][ST]  (k*g)^T        16KB
  _Float16* VT   = (_Float16*) (smem + 147456);   // [MEM][ST]  v^T            16KB
  _Float16* RD   = (_Float16*) (smem + 163840);   // [ST][MEM]  reads          16KB
  float*    GL   = (float*)    (smem + 180224);   // [ST] gates + [ST]=decay   512B
  _Float16* WQL  = (_Float16*) (smem + 180736);   // [D][D]   f16 Wq           32KB
  _Float16* WKL  = (_Float16*) (smem + 213504);   // [D][D]   f16 Wk           32KB
  _Float16* WVL  = (_Float16*) (smem + 246272);   // [MEM][D] f16 Wv           32KB
  _Float16* WOL  = (_Float16*) (smem + 279040);   // [D][MEM] f16 Wout         32KB
                                                  // total 311808 B (< 320KB WGP LDS)

  const int tid  = threadIdx.x;
  const int lane = tid & 31;
  const int wave = tid >> 5;
  const int b    = blockIdx.x;

  const float* xb   = x   + (size_t)b * SEQ * DIM;
  float*       outb = out + (size_t)b * SEQ * DIM;

  const int mm_base = (lane >> 4) << 3;   // C/D fragment row base (0 or 8)
  const int cfn     = lane & 15;          // C/D fragment column

  // ---- prologue: zero M, stage f16 weights in LDS -------------------------
  for (int i = tid; i < MEMD * DIM; i += 256) { MT32[i] = 0.0f; MT16[i] = (_Float16)0.0f; }
  for (int i = tid; i < DIM * DIM; i += 256) {
    WQL[i] = (_Float16)Wq[i];
    WKL[i] = (_Float16)Wk[i];
    WVL[i] = (_Float16)Wv[i];
    WOL[i] = (_Float16)Wout[i];
  }
  __syncthreads();

  for (int c = 0; c < NCHUNK; ++c) {
    // ---- prefetch next chunk's x (one cacheline per thread) --------------
    if (c + 1 < NCHUNK) {
      const float* xn = xb + (size_t)(c + 1) * ST * DIM;
      __builtin_prefetch(xn + (size_t)tid * 32, 0, 0);
    }

    // ---- load + convert current x chunk (ST x DIM) into LDS --------------
    const float* xc = xb + (size_t)c * ST * DIM;
    for (int i = tid; i < (ST * DIM) / 4; i += 256) {
      v4f xv = *((const v4f*)xc + i);
      v4h h;
#pragma unroll
      for (int j = 0; j < 4; ++j) h[j] = (_Float16)xv[j];
      *((v4h*)XL + i) = h;
    }
    __syncthreads();

    // ---- per-token gate g = sigmoid(x . Wg_w + b) -------------------------
    if (tid < ST) {
      float acc = Wg_b[0];
#pragma unroll 4
      for (int d = 0; d < DIM; ++d) acc += (float)XL[tid * DIM + d] * Wg_w[d];
      GL[tid] = 1.0f / (1.0f + __expf(-acc));
    }
    __syncthreads();
    if (tid == 0) {
      float s = 0.0f;
      for (int i = 0; i < ST; ++i) s += GL[i];
      GL[ST] = 1.0f - s * (1.0f / (float)ST);   // decay = 1 - avg_gate
    }

    // ---- projections: compile-time unrolled over q / k*g / v --------------
#pragma unroll
    for (int proj = 0; proj < 3; ++proj) {
      const _Float16* WB = (proj == 0) ? WQL : (proj == 1) ? WKL : WVL;
      for (int t = wave; t < 32; t += 8) {
        const int rt = t >> 3;           // token row-tile 0..3
        const int ct = t & 7;            // out-dim col-tile 0..7
        v8f acc = {};
#pragma unroll
        for (int kk = 0; kk < 4; ++kk) {
          v16h a  = load_frag_a (XL + rt * 16 * DIM + kk * 32, DIM, lane);
          v16h bt = load_frag_bT(WB + ct * 16 * DIM + kk * 32, DIM, lane);
          acc = __builtin_amdgcn_wmma_f32_16x16x32_f16(false, a, false, bt,
                                                       (short)0, acc, false, false);
        }
        if (proj == 0) {
#pragma unroll
          for (int e = 0; e < 8; ++e) {
            int sm = rt * 16 + mm_base + e;
            QL[sm * DIM + ct * 16 + cfn] = (_Float16)acc[e];
          }
        } else if (proj == 1) {
#pragma unroll
          for (int e = 0; e < 8; ++e) {
            int sm = rt * 16 + mm_base + e;
            KGT[(ct * 16 + cfn) * ST + sm] = (_Float16)(acc[e] * GL[sm]);
          }
        } else {
#pragma unroll
          for (int e = 0; e < 8; ++e) {
            int sm = rt * 16 + mm_base + e;
            VT[(ct * 16 + cfn) * ST + sm] = (_Float16)acc[e];
          }
        }
      }
    }
    __syncthreads();

    // ---- reads = q_chunk @ M  (M read BEFORE update) ----------------------
    for (int t = wave; t < 32; t += 8) {
      const int rt = t >> 3, ct = t & 7;
      v8f acc = {};
#pragma unroll
      for (int kk = 0; kk < 4; ++kk) {
        v16h a  = load_frag_a (QL   + rt * 16 * DIM + kk * 32, DIM, lane);
        v16h bt = load_frag_bT(MT16 + ct * 16 * DIM + kk * 32, DIM, lane);
        acc = __builtin_amdgcn_wmma_f32_16x16x32_f16(false, a, false, bt,
                                                     (short)0, acc, false, false);
      }
#pragma unroll
      for (int e = 0; e < 8; ++e)
        RD[(rt * 16 + mm_base + e) * MEMD + ct * 16 + cfn] = (_Float16)acc[e];
    }
    __syncthreads();

    const float decay = GL[ST];

    // ---- out = reads @ Wout^T  (independent of update) --------------------
    for (int t = wave; t < 32; t += 8) {
      const int rt = t >> 3, ct = t & 7;
      v8f acc = {};
#pragma unroll
      for (int kk = 0; kk < 4; ++kk) {
        v16h a  = load_frag_a (RD  + rt * 16 * MEMD + kk * 32, MEMD, lane);
        v16h bt = load_frag_bT(WOL + ct * 16 * MEMD + kk * 32, MEMD, lane);
        acc = __builtin_amdgcn_wmma_f32_16x16x32_f16(false, a, false, bt,
                                                     (short)0, acc, false, false);
      }
      float* og = outb + (size_t)(c * ST + rt * 16) * DIM + ct * 16;
#pragma unroll
      for (int e = 0; e < 8; ++e)
        og[(mm_base + e) * DIM + cfn] = acc[e];
    }

    // ---- M^T <- decay*M^T + v^T @ kg  (write^T) ---------------------------
    for (int t = wave; t < 64; t += 8) {
      const int rm = t >> 3;             // mem row-tile 0..7
      const int cd = t & 7;              // dim col-tile 0..7
      v8f acc;
#pragma unroll
      for (int e = 0; e < 8; ++e)
        acc[e] = decay * MT32[(rm * 16 + mm_base + e) * DIM + cd * 16 + cfn];
#pragma unroll
      for (int kk = 0; kk < 2; ++kk) {
        v16h a  = load_frag_a (VT  + rm * 16 * ST + kk * 32, ST, lane);
        v16h bt = load_frag_bT(KGT + cd * 16 * ST + kk * 32, ST, lane);
        acc = __builtin_amdgcn_wmma_f32_16x16x32_f16(false, a, false, bt,
                                                     (short)0, acc, false, false);
      }
#pragma unroll
      for (int e = 0; e < 8; ++e) {
        const int idx = (rm * 16 + mm_base + e) * DIM + cd * 16 + cfn;
        MT32[idx] = acc[e];
        MT16[idx] = (_Float16)acc[e];
      }
    }
    __syncthreads();
  }
}

// ---------------------------------------------------------------------------
extern "C" void kernel_launch(void* const* d_in, const int* in_sizes, int n_in,
                              void* d_out, int out_size, void* d_ws, size_t ws_size,
                              hipStream_t stream) {
  (void)in_sizes; (void)n_in; (void)out_size; (void)d_ws; (void)ws_size;

  const float* x    = (const float*)d_in[0];
  const float* Wq   = (const float*)d_in[1];
  const float* Wk   = (const float*)d_in[2];
  const float* Wv   = (const float*)d_in[3];
  const float* Wgw  = (const float*)d_in[4];
  const float* Wgb  = (const float*)d_in[5];
  const float* Wout = (const float*)d_in[6];
  float* out = (float*)d_out;

  const size_t smem = 311808;  // M^T(f32+f16) + chunk buffers + f16 weights
  nm_scan_kernel<<<BATCH, 256, smem, stream>>>(x, Wq, Wk, Wv, Wgw, Wgb, Wout, out);
}
